// SimpleLSTM_16252156248327
// MI455X (gfx1250) — matrix operand order repaired and checked by reading, not compiled, not run on hardware
//
#include <hip/hip_runtime.h>

typedef __attribute__((ext_vector_type(16))) _Float16 v16h;
typedef __attribute__((ext_vector_type(8)))  float    v8f;

#define NB 32          // batch
#define NS 128         // stocks
#define NT 256         // timesteps
#define NF 32          // input features
#define NH 64          // hidden
#define XH_STRIDE 104  // 96 (F+H) padded to avoid LDS bank conflicts

__device__ __forceinline__ v8f wmma_f16(v16h a, v16h b, v8f c) {
  // D(f32 16x16) = A(f16 16x32) * B(f16 32x16) + C
  return __builtin_amdgcn_wmma_f32_16x16x32_f16(false, a, false, b,
                                                (short)0, c, false, false);
}

// CDNA5 has a hardware transcendental tanh (v_tanh_f32).
__device__ __forceinline__ float fast_tanh(float x) {
#if __has_builtin(__builtin_amdgcn_tanhf)
  return __builtin_amdgcn_tanhf(x);
#elif __has_builtin(__builtin_amdgcn_tanh_f32)
  return __builtin_amdgcn_tanh_f32(x);
#else
  float r;
  asm volatile("v_tanh_f32 %0, %1\n\tv_nop" : "=v"(r) : "v"(x));
  return r;
#endif
}

// sigmoid(x) = 0.5 + 0.5 * tanh(0.5 x): 1 trans + 2 VALU, no branches
__device__ __forceinline__ float fast_sigmoid(float x) {
  return fmaf(0.5f, fast_tanh(0.5f * x), 0.5f);
}

__device__ __forceinline__ v16h load_bfrag16(const float* __restrict__ p) {
  v16h r;
#pragma unroll
  for (int i = 0; i < 16; ++i) r[i] = (_Float16)p[i];
  return r;
}

__global__ __launch_bounds__(256) void lstm_stock_kernel(
    const float* __restrict__ features,
    const float* __restrict__ W_ih, const float* __restrict__ W_hh,
    const float* __restrict__ b_ih, const float* __restrict__ b_hh,
    const float* __restrict__ W_out, const float* __restrict__ b_out,
    float* __restrict__ out)
{
  const int s    = blockIdx.x;        // one stock per block
  const int tid  = threadIdx.x;       // 256 threads = 8 wave32
  const int lane = tid & 31;
  const int wave = tid >> 5;
  const int mt   = wave & 1;          // M tile (batch rows mt*16..mt*16+15)
  const int ng   = wave >> 1;         // N group: gate tiles g*64 + ng*16
  const int l15  = lane & 15;
  const int lhi  = lane >> 4;

  __shared__ _Float16 xh[NB][XH_STRIDE];  // [batch][ x(0..31) | h(32..95) ]

  // zero LDS so h starts at 0
  for (int i = tid; i < NB * XH_STRIDE; i += 256)
    ((_Float16*)xh)[i] = (_Float16)0.0f;

  // per-gate bias broadcast as WMMA C-operand init (bias depends only on N=lane)
  v8f biasv[4];
#pragma unroll
  for (int g = 0; g < 4; ++g) {
    int col = g * NH + ng * 16 + l15;
    float bv = b_ih[s * 4 * NH + col] + b_hh[s * 4 * NH + col];
#pragma unroll
    for (int e = 0; e < 8; ++e) biasv[g][e] = bv;
  }

  // Loop-invariant B fragments (f16) kept in VGPRs.
  // B k order used here (contiguous, same as A): lane<16 -> col=lane, K=0..15; lane>=16 -> K=16..31.
  // gates = A * W^T, so B[k][n] = W[n][k]: 16 contiguous floats of W row n.
  v16h bfrag[4][3];
  {
    const float* Wih_s = W_ih + (size_t)s * 4 * NH * NF;
    const float* Whh_s = W_hh + (size_t)s * 4 * NH * NH;
    const int hi16 = lhi * 16;
#pragma unroll
    for (int g = 0; g < 4; ++g) {
      int n = g * NH + ng * 16 + l15;
      bfrag[g][0] = load_bfrag16(Wih_s + n * NF + hi16);        // K  0..31 (x)
      bfrag[g][1] = load_bfrag16(Whh_s + n * NH + hi16);        // K 32..63 (h lo)
      bfrag[g][2] = load_bfrag16(Whh_s + n * NH + 32 + hi16);   // K 64..95 (h hi)
    }
  }

  // Per-thread x_t slice: thread -> (batch row, 4 features)
  const int xb = tid >> 3;
  const int xf = (tid & 7) * 4;
  const float* fptr = features + (((size_t)xb * NS + s) * NT) * NF + xf;

  // stage x_0 into LDS
  {
    float4 x0 = *(const float4*)(fptr);
    union { _Float16 h[4]; uint2 u; } cvx;
    cvx.h[0] = (_Float16)x0.x; cvx.h[1] = (_Float16)x0.y;
    cvx.h[2] = (_Float16)x0.z; cvx.h[3] = (_Float16)x0.w;
    *(uint2*)&xh[xb][xf] = cvx.u;
  }

  // lane-private cell state: element e -> row mt*16 + 8*lhi + e, col ng*16+l15
  float cst[8];
#pragma unroll
  for (int e = 0; e < 8; ++e) cst[e] = 0.0f;

  __syncthreads();

  const int arow  = mt * 16 + l15;   // A row for this lane
  const int mbase = mt * 16 + 8 * lhi;
  const int hc    = ng * 16 + l15;   // owned h column

  for (int t = 0; t < NT; ++t) {
    // A fragments (contiguous k order, as in B: lane<16 K=0..15; lane>=16 K=16..31)
    v16h afrag[3];
#pragma unroll
    for (int ks = 0; ks < 3; ++ks) {
      const _Float16* ap = &xh[arow][ks * 32 + 16 * lhi];
      v16h a;
#pragma unroll
      for (int i = 0; i < 8; ++i) a[i] = ap[i];
#pragma unroll
      for (int i = 0; i < 8; ++i) a[8 + i] = ap[8 + i];
      afrag[ks] = a;
    }

    __syncthreads();  // order LDS A-reads before LDS h/x writes (only dscnt)

    // Issue next-x global load AFTER the barrier so its s_wait sinks to the
    // ds_store at the end of the step (overlaps with the 12 WMMAs below).
    float4 xnext;
    const bool more = (t + 1 < NT);
    if (more) xnext = *(const float4*)(fptr + (size_t)(t + 1) * NF);
    if (t + 2 < NT) __builtin_prefetch(fptr + (size_t)(t + 2) * NF, 0, 3);

    // gates[m][n] over K = [x | h]; accumulators start from broadcast bias
    v8f acc[4];
#pragma unroll
    for (int g = 0; g < 4; ++g) {
      v8f c = wmma_f16(afrag[0], bfrag[g][0], biasv[g]);
      c     = wmma_f16(afrag[1], bfrag[g][1], c);
      acc[g] = wmma_f16(afrag[2], bfrag[g][2], c);
    }

    // LSTM cell update: branch-free hw-tanh activations, fp32 state
#pragma unroll
    for (int e = 0; e < 8; ++e) {
      float gi = fast_sigmoid(acc[0][e]);
      float gf = fast_sigmoid(acc[1][e]);
      float gg = fast_tanh   (acc[2][e]);
      float go = fast_sigmoid(acc[3][e]);
      float cn = fmaf(gf, cst[e], gi * gg);
      cst[e] = cn;
      xh[mbase + e][NF + hc] = (_Float16)(go * fast_tanh(cn));
    }

    if (more) {
      union { _Float16 h[4]; uint2 u; } cvx;
      cvx.h[0] = (_Float16)xnext.x; cvx.h[1] = (_Float16)xnext.y;
      cvx.h[2] = (_Float16)xnext.z; cvx.h[3] = (_Float16)xnext.w;
      *(uint2*)&xh[xb][xf] = cvx.u;
    }
    __syncthreads();  // h/x stores visible before next iteration's A reads
  }

  // per-stock output head: out[b,s] = h_last[b,:] . W_out[s,0,:] + b_out[s]
  if (tid < NB) {
    float acc = b_out[s];
    const float* wo = W_out + s * NH;
#pragma unroll
    for (int h = 0; h < NH; ++h)
      acc += (float)xh[tid][NF + h] * wo[h];
    out[tid * NS + s] = acc;
  }
}

extern "C" void kernel_launch(void* const* d_in, const int* in_sizes, int n_in,
                              void* d_out, int out_size, void* d_ws, size_t ws_size,
                              hipStream_t stream) {
  const float* features = (const float*)d_in[0];
  const float* W_ih     = (const float*)d_in[1];
  const float* W_hh     = (const float*)d_in[2];
  const float* b_ih     = (const float*)d_in[3];
  const float* b_hh     = (const float*)d_in[4];
  const float* W_out    = (const float*)d_in[5];
  const float* b_out    = (const float*)d_in[6];
  float* out = (float*)d_out;
  (void)in_sizes; (void)n_in; (void)out_size; (void)d_ws; (void)ws_size;
  lstm_stock_kernel<<<NS, 256, 0, stream>>>(features, W_ih, W_hh,
                                            b_ih, b_hh, W_out, b_out, out);
}
